// EvoformerBlock_19241453486250
// MI455X (gfx1250) — matrix operand order repaired and checked by reading, not compiled, not run on hardware
//
#include <hip/hip_runtime.h>
#include <stdint.h>

typedef __bf16 bf16;
typedef __attribute__((ext_vector_type(16))) __bf16 v16bf;
typedef __attribute__((ext_vector_type(8)))  float  v8f;
typedef int v4i128 __attribute__((vector_size(16)));   // matches builtin param type

#define DEV __device__ __forceinline__

DEV bf16 f2bf(float f) {
  union { float f; uint32_t u; } a; a.f = f;
  uint32_t r = (a.u + 0x7FFFu + ((a.u >> 16) & 1u)) >> 16;
  union { uint16_t s; bf16 b; } o; o.s = (uint16_t)r;
  return o.b;
}
DEV float sigm(float v) { return 1.f / (1.f + __expf(-v)); }

union FragU { v16bf v; uint4 q[2]; };

// A fragment (16x32 MxK), tile row-major with leading dim ldk (bf16 elems).
// ISA 7.12.2: lane<16 -> row=lane, K {0..7,16..23}; lane>=16 -> row=lane-16, K {8..15,24..31}
DEV v16bf frag_a(const bf16* tile, int row0, int ldk) {
  int lane = threadIdx.x & 31;
  const bf16* p = tile + (size_t)(row0 + (lane & 15)) * ldk + ((lane >> 4) << 3);
  FragU f;
  f.q[0] = *(const uint4*)(p);
  f.q[1] = *(const uint4*)(p + 16);
  return f.v;
}
// B fragment (32x16 KxN) from an N-major tile (tileT[n][k], leading dim ldk).
// lane<16 -> col=lane, K {0..7,16..23}; lane>=16 -> col=lane-16, K {8..15,24..31}
DEV v16bf frag_b(const bf16* tileT, int col0, int ldk) {
  int lane = threadIdx.x & 31;
  const bf16* p = tileT + (size_t)(col0 + (lane & 15)) * ldk + ((lane >> 4) << 3);
  FragU f;
  f.q[0] = *(const uint4*)(p);
  f.q[1] = *(const uint4*)(p + 16);
  return f.v;
}

#define WMMA_BF16(A_, B_, C_) \
  __builtin_amdgcn_wmma_f32_16x16x32_bf16(false, (A_), false, (B_), (short)0, (C_), false, false)

// --- CDNA5 async global->LDS copy (ASYNCcnt-tracked); sync fallback ---------
#if __has_builtin(__builtin_amdgcn_global_load_async_to_lds_b128) && \
    __has_builtin(__builtin_amdgcn_s_wait_asynccnt)
#define HAS_ASYNC 1
#else
#define HAS_ASYNC 0
#endif

DEV void cp_b128(bf16* ldst, const bf16* gsrc) {
#if HAS_ASYNC
  __builtin_amdgcn_global_load_async_to_lds_b128(
      (v4i128*)const_cast<bf16*>(gsrc),
      (__attribute__((address_space(3))) v4i128*)ldst, 0, 0);
#else
  *(uint4*)ldst = *(const uint4*)gsrc;
#endif
}
DEV void wait_async_lds() {
#if HAS_ASYNC
  __builtin_amdgcn_s_wait_asynccnt(0);
#endif
}

// ---------------------------------------------------------------------------
// Generic batched GEMM: C[b] = act(A[b] (MxK, row-major) * Bt[b]^T + bias)
// Bt stored as [N][K] (pre-transposed). f32 and/or bf16 outputs.
// act: 0 none, 1 relu, 2 sigmoid. M multiple of 128; K multiple of 32; N guarded.
// Block tile 128x64, 8 waves each owning a 32x32 subtile (4 WMMA / k-step),
// double-buffered LDS filled with async-to-LDS copies.
// ---------------------------------------------------------------------------
__global__ __launch_bounds__(256) void k_gemm(
    const bf16* __restrict__ Ag, long sA, const bf16* __restrict__ Btg, long sB,
    float* __restrict__ Cg, long sC, bf16* __restrict__ Cbg, long sCb,
    const float* __restrict__ bias, int M, int N, int K, int act)
{
  __shared__ bf16 As[2][128 * 32];
  __shared__ bf16 Bs[2][64 * 32];
  const bf16* A  = Ag  + (size_t)blockIdx.z * sA;
  const bf16* Bt = Btg + (size_t)blockIdx.z * sB;
  float* C  = Cg  ? Cg  + (size_t)blockIdx.z * sC  : (float*)0;
  bf16*  Cb = Cbg ? Cbg + (size_t)blockIdx.z * sCb : (bf16*)0;
  int m0 = blockIdx.y * 128, n0 = blockIdx.x * 64;
  int t = threadIdx.x, lane = t & 31, w = t >> 5;
  int wm = (w >> 1) * 32, wn = (w & 1) * 32;   // 8 waves: 4x2 grid of 32x32 subtiles
  int lr = t >> 2, lc = (t & 3) * 8;           // cooperative tile fill (16B per op)
  int nr = n0 + lr;
  bool bok = nr < N;
  if (!bok) {                                   // zero-fill OOB B rows once
    uint4 zz; zz.x = 0u; zz.y = 0u; zz.z = 0u; zz.w = 0u;
    *(uint4*)&Bs[0][lr * 32 + lc] = zz;
    *(uint4*)&Bs[1][lr * 32 + lc] = zz;
  }
  auto fill = [&](int bb, int kk) {
    cp_b128(&As[bb][lr * 32 + lc],        &A[(size_t)(m0 + lr) * K + kk + lc]);
    cp_b128(&As[bb][(lr + 64) * 32 + lc], &A[(size_t)(m0 + lr + 64) * K + kk + lc]);
    if (bok) cp_b128(&Bs[bb][lr * 32 + lc], &Bt[(size_t)nr * K + kk + lc]);
    if (kk + 32 < K)
      __builtin_prefetch(&A[(size_t)(m0 + lr) * K + kk + 32 + lc], 0, 1);
  };
  v8f acc[2][2] = {};
  fill(0, 0);
  wait_async_lds();
  __syncthreads();
  int buf = 0;
  for (int k0 = 0; k0 < K; k0 += 32) {
    if (k0 + 32 < K) fill(buf ^ 1, k0 + 32);    // async prefetch of next tile
    v16bf a0 = frag_a(As[buf], wm, 32);
    v16bf a1 = frag_a(As[buf], wm + 16, 32);
    v16bf b0 = frag_b(Bs[buf], wn, 32);
    v16bf b1 = frag_b(Bs[buf], wn + 16, 32);
    acc[0][0] = WMMA_BF16(a0, b0, acc[0][0]);
    acc[0][1] = WMMA_BF16(a0, b1, acc[0][1]);
    acc[1][0] = WMMA_BF16(a1, b0, acc[1][0]);
    acc[1][1] = WMMA_BF16(a1, b1, acc[1][1]);
    wait_async_lds();                            // next tile landed in LDS
    __syncthreads();
    buf ^= 1;
  }
  int hi = lane >> 4, nlo = lane & 15;
#pragma unroll
  for (int rm = 0; rm < 2; ++rm) {
#pragma unroll
    for (int r = 0; r < 8; ++r) {
      int mm = m0 + wm + rm * 16 + r + hi * 8;
#pragma unroll
      for (int tb = 0; tb < 2; ++tb) {
        int nn = n0 + wn + tb * 16 + nlo;
        if (nn >= N) continue;
        float v = acc[rm][tb][r];
        if (bias) v += bias[nn];
        if (act == 1) v = fmaxf(v, 0.f);
        else if (act == 2) v = sigm(v);
        if (C)  C[(size_t)mm * N + nn] = v;
        if (Cb) Cb[(size_t)mm * N + nn] = f2bf(v);
      }
    }
  }
}

// ---------------------------------------------------------------------------
// Fused attention: per (batch b, head h, 32-query block).
// Q/K/V bf16 [B][256][H*32]; bias f32 [H][256][256]; kb f32 [B][256].
// O f32 [B][256][H*32].  S kept fully in LDS (256 keys), then softmax, then P*V.
// ---------------------------------------------------------------------------
__global__ __launch_bounds__(128) void k_attn(
    const bf16* __restrict__ Q, const bf16* __restrict__ Kg, const bf16* __restrict__ V,
    const float* __restrict__ bias, const float* __restrict__ kb,
    float* __restrict__ O, int H, float scale)
{
  extern __shared__ char smem[];
  bf16*  Qs = (bf16*)smem;             // 32 x 32
  bf16*  Ks = Qs + 32 * 32;            // 256 x 32  (key-major)
  bf16*  Vt = Ks + 256 * 32;           // 32 x 256  (ch-major = transposed V)
  float* S  = (float*)(Vt + 32 * 256); // 32 x 256
  bf16*  P  = (bf16*)(S + 32 * 256);   // 32 x 256
  int b = blockIdx.z, h = blockIdx.y, q0 = blockIdx.x * 32;
  int HC = H * 32;
  size_t bs = (size_t)256 * HC;
  const bf16* Qb = Q  + (size_t)b * bs + h * 32;
  const bf16* Kb = Kg + (size_t)b * bs + h * 32;
  const bf16* Vb = V  + (size_t)b * bs + h * 32;
  int t = threadIdx.x, lane = t & 31, w = t >> 5;
  { int row = t >> 2, c8 = (t & 3) * 8;
    *(uint4*)&Qs[row * 32 + c8] = *(const uint4*)&Qb[(size_t)(q0 + row) * HC + c8]; }
  for (int key = t; key < 256; key += 128) {
#pragma unroll
    for (int c8 = 0; c8 < 4; ++c8)
      *(uint4*)&Ks[key * 32 + c8 * 8] = *(const uint4*)&Kb[(size_t)key * HC + c8 * 8];
    for (int c = 0; c < 32; ++c) Vt[c * 256 + key] = Vb[(size_t)key * HC + c];
  }
  __syncthreads();
  // S = scale*Q.K^T + bias + kbias ; wave w: query half (w&1), key half (w>>1)
  {
    int qh = (w & 1) * 16, kh = (w >> 1) * 128;
    v16bf aq = frag_a(Qs, qh, 32);
    int hi = lane >> 4, nlo = lane & 15;
    for (int kt = 0; kt < 8; ++kt) {
      int kc = kh + kt * 16;
      v16bf bk = frag_b(Ks, kc, 32);
      v8f acc = {};
      acc = WMMA_BF16(aq, bk, acc);
#pragma unroll
      for (int r = 0; r < 8; ++r) {
        int row = qh + r + hi * 8;
        int col = kc + nlo;
        S[row * 256 + col] = acc[r] * scale
            + bias[((size_t)h * 256 + (q0 + row)) * 256 + col]
            + kb[(size_t)b * 256 + col];
      }
    }
  }
  __syncthreads();
  if (t < 32) {   // row softmax (one lane per query row)
    float mx = -3.0e38f;
    for (int k = 0; k < 256; ++k) mx = fmaxf(mx, S[t * 256 + k]);
    float sum = 0.f;
    for (int k = 0; k < 256; ++k) { float e = __expf(S[t * 256 + k] - mx); S[t * 256 + k] = e; sum += e; }
    float inv = 1.f / sum;
    for (int k = 0; k < 256; ++k) P[t * 256 + k] = f2bf(S[t * 256 + k] * inv);
  }
  __syncthreads();
  // O = P @ V ; wave w: query half (w&1), ch half (w>>1)
  {
    int qh = (w & 1) * 16, ch = (w >> 1) * 16;
    v8f acc = {};
    for (int kk = 0; kk < 8; ++kk) {
      v16bf ap = frag_a(P + kk * 32, qh, 256);
      v16bf bv = frag_b(Vt + kk * 32, ch, 256);
      acc = WMMA_BF16(ap, bv, acc);
    }
    int hi = lane >> 4, nlo = lane & 15;
    float* Ob = O + (size_t)b * bs + h * 32;
#pragma unroll
    for (int r = 0; r < 8; ++r) {
      int row = q0 + qh + r + hi * 8;
      Ob[(size_t)row * HC + ch + nlo] = acc[r];
    }
  }
}

// ---------------------------------------------------------------------------
// LayerNorm: one wave32 per row. Emits f32 and/or bf16.
// ---------------------------------------------------------------------------
__global__ __launch_bounds__(256) void k_ln(
    const float* __restrict__ X, const float* __restrict__ g, const float* __restrict__ b,
    float* __restrict__ Yf, bf16* __restrict__ Yb, int M, int D)
{
  int row = blockIdx.x * 8 + (threadIdx.x >> 5);
  if (row >= M) return;
  int lane = threadIdx.x & 31;
  const float* x = X + (size_t)row * D;
  float s = 0.f, s2 = 0.f;
  for (int i = lane; i < D; i += 32) { float v = x[i]; s += v; s2 += v * v; }
  for (int m = 16; m; m >>= 1) { s += __shfl_xor(s, m, 32); s2 += __shfl_xor(s2, m, 32); }
  float mu = s / D;
  float var = s2 / D - mu * mu;
  float r = rsqrtf(var + 1e-5f);
  for (int i = lane; i < D; i += 32) {
    float v = (x[i] - mu) * r * g[i] + b[i];
    if (Yf) Yf[(size_t)row * D + i] = v;
    if (Yb) Yb[(size_t)row * D + i] = f2bf(v);
  }
}

// ------------------------- elementwise / repack kernels ---------------------
#define EWIDX size_t i = (size_t)blockIdx.x * 256 + threadIdx.x; if (i >= n) return;

__global__ void k_cvt_wt(const float* __restrict__ W, bf16* __restrict__ Wt,
                         int K, int N, size_t n) {            // Wt[n][k] = W[k][n]
  EWIDX int nn = (int)(i / K), k = (int)(i % K);
  Wt[i] = f2bf(W[(size_t)k * N + nn]);
}
__global__ void k_add(float* d, const float* s, size_t n) { EWIDX d[i] += s[i]; }
__global__ void k_add_rowmask(float* d, const float* s, const float* rm, int C, size_t n) {
  EWIDX d[i] += s[i] * rm[i / C];
}
__global__ void k_gate_bf(bf16* o, const float* p, const float* g, size_t n) {
  EWIDX o[i] = f2bf(p[i] * sigm(g[i]));
}
__global__ void k_gatemask(float* o, const float* p, const float* g, const float* rm, int C, size_t n) {
  EWIDX o[i] = p[i] * sigm(g[i]) * rm[i / C];
}
__global__ void k_combine(float* d, const float* a, const float* g, size_t n) {
  EWIDX d[i] += a[i] * sigm(g[i]);
}
__global__ void k_rowmask_cvt(const float* s, const float* rm, bf16* o, int C, size_t n) {
  EWIDX o[i] = f2bf(s[i] * rm[i / C]);
}
__global__ void k_pairmask(const float* mask, float* pm, size_t n) {
  EWIDX pm[i] = mask[i >> 8] * mask[i & 255];
}
__global__ void k_kbias(const float* s, float* d, size_t n) { EWIDX d[i] = 1.0e9f * (s[i] - 1.f); }
__global__ void k_bias_build(float* bias, const float* raw, int H, size_t n) {
  EWIDX int h = (int)(i >> 16); size_t rc = i & 65535;
  bias[i] = raw[rc * H + h];
}
__global__ void k_pack(const float* src, bf16* dst, int tr, size_t n) {  // dst[c][x][y]
  EWIDX int c = (int)(i >> 16); int xy = (int)(i & 65535);
  int x = xy >> 8, y = xy & 255;
  size_t s = ((size_t)(tr ? y * 256 + x : x * 256 + y)) * 128 + c;
  dst[i] = f2bf(src[s]);
}
__global__ void k_unpack(const float* src, float* dst, size_t n) {       // [c][ij] -> [ij][c]
  EWIDX size_t row = i >> 7; int c = (int)(i & 127);
  dst[i] = src[(size_t)c * 65536 + row];
}
__global__ void k_transpose_z(const float* src, float* dst, size_t n) {  // z[i][j][c] -> z[j][i][c]
  EWIDX size_t row = i >> 7; int c = (int)(i & 127);
  int ii = (int)(row >> 8), j = (int)(row & 255);
  dst[((size_t)(j * 256 + ii)) * 128 + c] = src[i];
}
__global__ void k_add_T(float* d, const float* s, size_t n) {            // d[i][j][c] += s[j][i][c]
  EWIDX size_t row = i >> 7; int c = (int)(i & 127);
  int ii = (int)(row >> 8), j = (int)(row & 255);
  d[i] += s[((size_t)(j * 256 + ii)) * 128 + c];
}
__global__ void k_transpose_nn(const float* s, float* d, size_t n) {
  EWIDX int ii = (int)(i >> 8), j = (int)(i & 255);
  d[(size_t)j * 256 + ii] = s[i];
}
__global__ void k_opm_final(float* z, const float* o, const float* pm, size_t n) {
  EWIDX z[i] += o[i] / (0.001f + pm[i >> 7]);
}
__global__ void k_prep_opwo(const float* wo, bf16* wt, size_t n) { // [(d*128+z)][c] <- wo[(c*32+d)][z]
  EWIDX size_t row = i >> 5; int c = (int)(i & 31);
  int d = (int)(row >> 7), zc = (int)(row & 127);
  wt[i] = f2bf(wo[((size_t)c * 32 + d) * 128 + zc]);
}
__global__ void k_prep_T(const float* T, bf16* Tt, size_t n) {     // Tt[i][z*32+d] <- T[i][d*128+z]
  EWIDX size_t row = i >> 12; int r = (int)(i & 4095);
  int zc = r >> 5, d = r & 31;
  Tt[i] = f2bf(T[(row << 12) + (size_t)d * 128 + zc]);
}

#define EW(n) dim3((unsigned)(((n) + 255) / 256)), 256, 0, stream

// ---------------------------------------------------------------------------
extern "C" void kernel_launch(void* const* d_in, const int* in_sizes, int n_in,
                              void* d_out, int out_size, void* d_ws, size_t ws_size,
                              hipStream_t stream) {
  (void)in_sizes; (void)n_in; (void)out_size; (void)ws_size;
  const size_t ZT = (size_t)256 * 256 * 128;   // 8388608
  const size_t MT = (size_t)256 * 384;         // 98304
  const float* m_in = (const float*)d_in[0];
  const float* z_in = (const float*)d_in[1];
  const float* mask = (const float*)d_in[2];
  int pi = 3;
  auto nxt = [&]() { return (const float*)d_in[pi++]; };
  // --- params in _make_params() insertion order ---
  const float *mra_ln_m_g = nxt(), *mra_ln_m_b = nxt();
  const float *mra_ln_z_g = nxt(), *mra_ln_z_b = nxt();
  const float *mra_wb = nxt();
  const float *mra_wq = nxt(), *mra_wk = nxt(), *mra_wv = nxt(), *mra_wgp = nxt();
  const float *mra_bg = nxt();
  const float *mra_wo = nxt(), *mra_bo = nxt();
  const float *mt_ln_g = nxt(), *mt_ln_b = nxt();
  const float *mt_w1 = nxt(), *mt_b1 = nxt(), *mt_w2 = nxt(), *mt_b2 = nxt();
  const float *opm_ln_g = nxt(), *opm_ln_b = nxt();
  const float *opm_w1 = nxt(), *opm_b1 = nxt(), *opm_w2 = nxt(), *opm_b2 = nxt();
  const float *opm_wo = nxt(), *opm_bo = nxt();
  struct TM { const float *lng,*lnb,*wap,*bap,*wag,*bag,*wbp,*bbp,*wbg,*bbg,*log,*lob,*wz,*bz,*wg,*bg2; } tm[2];
  for (int t = 0; t < 2; ++t) {
    tm[t].lng = nxt(); tm[t].lnb = nxt();
    tm[t].wap = nxt(); tm[t].bap = nxt(); tm[t].wag = nxt(); tm[t].bag = nxt();
    tm[t].wbp = nxt(); tm[t].bbp = nxt(); tm[t].wbg = nxt(); tm[t].bbg = nxt();
    tm[t].log = nxt(); tm[t].lob = nxt();
    tm[t].wz = nxt();  tm[t].bz = nxt();  tm[t].wg = nxt();  tm[t].bg2 = nxt();
  }
  struct TA { const float *lng,*lnb,*wq,*wk,*wv,*wg,*wb,*bg,*wo,*bo; } ta[2];
  for (int t = 0; t < 2; ++t) {
    ta[t].lng = nxt(); ta[t].lnb = nxt();
    ta[t].wq = nxt(); ta[t].wk = nxt(); ta[t].wv = nxt(); ta[t].wg = nxt();
    ta[t].wb = nxt(); ta[t].bg = nxt(); ta[t].wo = nxt(); ta[t].bo = nxt();
  }
  const float *pt_ln_g = nxt(), *pt_ln_b = nxt();
  const float *pt_w1 = nxt(), *pt_b1 = nxt(), *pt_w2 = nxt(), *pt_b2 = nxt();

  // --- workspace carve ---
  char* base = (char*)d_ws; size_t off = 0;
  auto alc = [&](size_t bytes) { void* p = base + off; off += (bytes + 255) & ~(size_t)255; return p; };
  float* mW  = (float*)alc(MT * 4);
  float* zW  = (float*)alc(ZT * 4);
  float* F0  = (float*)alc(ZT * 4);
  float* F1  = (float*)alc(ZT * 4);
  float* F2  = (float*)alc(ZT * 4);
  float* E0  = (float*)alc(ZT * 4);      // einsum out / zT
  bf16*  xbf = (bf16*)alc(ZT * 2);
  bf16*  hbf = (bf16*)alc((size_t)65536 * 512 * 2);
  bf16*  B0  = (bf16*)alc(ZT * 2);
  bf16*  B1  = (bf16*)alc(ZT * 2);
  bf16*  B2  = (bf16*)alc(ZT * 2);
  bf16*  B3  = (bf16*)alc(ZT * 2);
  bf16*  B4  = (bf16*)alc(ZT * 2);
  float* biasb = (float*)alc((size_t)8 * 65536 * 4);
  float* kb    = (float*)alc((size_t)65536 * 4);
  float* pm    = (float*)alc((size_t)65536 * 4);
  float* pmT   = (float*)alc((size_t)65536 * 4);
  bf16* wt_mra[4]; for (int i = 0; i < 4; ++i) wt_mra[i] = (bf16*)alc((size_t)256 * 384 * 2);
  bf16* wt_mrao = (bf16*)alc((size_t)384 * 256 * 2);
  bf16* wt_mrab = (bf16*)alc((size_t)8 * 128 * 2);
  bf16* wt_mt1  = (bf16*)alc((size_t)1536 * 384 * 2);
  bf16* wt_mt2  = (bf16*)alc((size_t)384 * 1536 * 2);
  bf16* wt_op1  = (bf16*)alc((size_t)32 * 384 * 2);
  bf16* wt_op2  = (bf16*)alc((size_t)32 * 384 * 2);
  bf16* wt_opwo = (bf16*)alc((size_t)4096 * 32 * 2);
  bf16* wt_tm[2][6];
  for (int t = 0; t < 2; ++t) for (int i = 0; i < 6; ++i) wt_tm[t][i] = (bf16*)alc((size_t)128 * 128 * 2);
  bf16* wt_ta[2][5]; bf16* wt_tab[2];
  for (int t = 0; t < 2; ++t) {
    for (int i = 0; i < 5; ++i) wt_ta[t][i] = (bf16*)alc((size_t)128 * 128 * 2);
    wt_tab[t] = (bf16*)alc((size_t)4 * 128 * 2);
  }
  bf16* wt_pt1 = (bf16*)alc((size_t)512 * 128 * 2);
  bf16* wt_pt2 = (bf16*)alc((size_t)128 * 512 * 2);

  // --- helpers ---
  auto gemm = [&](const bf16* A_, long sA, const bf16* Bt_, long sB,
                  float* C_, long sC, bf16* Cb_, long sCb, const float* bias_,
                  int M, int N, int K, int act, int batch) {
    dim3 g((unsigned)((N + 63) / 64), (unsigned)((M + 127) / 128), (unsigned)batch);
    k_gemm<<<g, 256, 0, stream>>>(A_, sA, Bt_, sB, C_, sC, Cb_, sCb, bias_, M, N, K, act);
  };
  auto ln = [&](const float* X, const float* g_, const float* b_, float* Yf, bf16* Yb, int M, int D) {
    k_ln<<<dim3((unsigned)((M + 7) / 8)), 256, 0, stream>>>(X, g_, b_, Yf, Yb, M, D);
  };
  auto attn = [&](const bf16* Q, const bf16* K, const bf16* V, const float* bias_,
                  const float* kb_, float* O, int H, int B) {
    dim3 g(8, (unsigned)H, (unsigned)B);
    k_attn<<<g, 128, 83968, stream>>>(Q, K, V, bias_, kb_, O, H, 0.17677669529663687f);
  };
  auto cvtW = [&](const float* W, bf16* Wt, int K, int N) {
    size_t n = (size_t)K * N;
    k_cvt_wt<<<EW(n)>>>(W, Wt, K, N, n);
  };

  // --- working copies + masks ---
  (void)hipMemcpyAsync(mW, m_in, MT * 4, hipMemcpyDeviceToDevice, stream);
  (void)hipMemcpyAsync(zW, z_in, ZT * 4, hipMemcpyDeviceToDevice, stream);
  k_pairmask<<<EW(65536)>>>(mask, pm, 65536);
  k_transpose_nn<<<EW(65536)>>>(pm, pmT, 65536);

  // --- weight convert+transpose to [N][K] bf16 ---
  cvtW(mra_wq, wt_mra[0], 384, 256); cvtW(mra_wk, wt_mra[1], 384, 256);
  cvtW(mra_wv, wt_mra[2], 384, 256); cvtW(mra_wgp, wt_mra[3], 384, 256);
  cvtW(mra_wo, wt_mrao, 256, 384);
  cvtW(mra_wb, wt_mrab, 128, 8);
  cvtW(mt_w1, wt_mt1, 384, 1536); cvtW(mt_w2, wt_mt2, 1536, 384);
  cvtW(opm_w1, wt_op1, 384, 32);  cvtW(opm_w2, wt_op2, 384, 32);
  { size_t n = (size_t)4096 * 32; k_prep_opwo<<<EW(n)>>>(opm_wo, wt_opwo, n); }
  for (int t = 0; t < 2; ++t) {
    const float* ws6[6] = { tm[t].wap, tm[t].wag, tm[t].wbp, tm[t].wbg, tm[t].wz, tm[t].wg };
    for (int i = 0; i < 6; ++i) cvtW(ws6[i], wt_tm[t][i], 128, 128);
    const float* wa5[5] = { ta[t].wq, ta[t].wk, ta[t].wv, ta[t].wg, ta[t].wo };
    for (int i = 0; i < 5; ++i) cvtW(wa5[i], wt_ta[t][i], 128, 128);
    cvtW(ta[t].wb, wt_tab[t], 128, 4);
  }
  cvtW(pt_w1, wt_pt1, 128, 512); cvtW(pt_w2, wt_pt2, 512, 128);

  // ======================= 1) MSA row attention =======================
  ln(mW, mra_ln_m_g, mra_ln_m_b, (float*)0, xbf, 256, 384);
  ln(zW, mra_ln_z_g, mra_ln_z_b, (float*)0, B4, 65536, 128);          // zl
  gemm(B4, 0, wt_mrab, 0, F2, 0, (bf16*)0, 0, (const float*)0, 65536, 8, 128, 0, 1); // pb raw
  k_bias_build<<<EW((size_t)8 * 65536)>>>(biasb, F2, 8, (size_t)8 * 65536);
  k_kbias<<<EW(256)>>>(mask, kb, 256);
  gemm(xbf, 0, wt_mra[0], 0, (float*)0, 0, B0, 0, (const float*)0, 256, 256, 384, 0, 1); // q
  gemm(xbf, 0, wt_mra[1], 0, (float*)0, 0, B1, 0, (const float*)0, 256, 256, 384, 0, 1); // k
  gemm(xbf, 0, wt_mra[2], 0, (float*)0, 0, B2, 0, (const float*)0, 256, 256, 384, 0, 1); // v
  gemm(xbf, 0, wt_mra[3], 0, F1, 0, (bf16*)0, 0, mra_bg, 256, 256, 384, 0, 1);          // gate logits
  attn(B0, B1, B2, biasb, kb, F0, 8, 1);
  k_gate_bf<<<EW((size_t)256 * 256)>>>(B3, F0, F1, (size_t)256 * 256);
  gemm(B3, 0, wt_mrao, 0, F2, 0, (bf16*)0, 0, mra_bo, 256, 384, 256, 0, 1);
  k_add<<<EW(MT)>>>(mW, F2, MT);

  // ======================= 2) MSA transition =======================
  ln(mW, mt_ln_g, mt_ln_b, (float*)0, xbf, 256, 384);
  gemm(xbf, 0, wt_mt1, 0, (float*)0, 0, hbf, 0, mt_b1, 256, 1536, 384, 1, 1);
  gemm(hbf, 0, wt_mt2, 0, F0, 0, (bf16*)0, 0, mt_b2, 256, 384, 1536, 0, 1);
  k_add_rowmask<<<EW(MT)>>>(mW, F0, mask, 384, MT);

  // ======================= 3) Outer product mean =======================
  ln(mW, opm_ln_g, opm_ln_b, (float*)0, xbf, 256, 384);
  gemm(xbf, 0, wt_op1, 0, F0, 0, (bf16*)0, 0, opm_b1, 256, 32, 384, 0, 1);
  k_rowmask_cvt<<<EW((size_t)256 * 32)>>>(F0, mask, B0, 32, (size_t)256 * 32);       // a
  gemm(xbf, 0, wt_op2, 0, F0, 0, (bf16*)0, 0, opm_b2, 256, 32, 384, 0, 1);
  k_rowmask_cvt<<<EW((size_t)256 * 32)>>>(F0, mask, B1, 32, (size_t)256 * 32);       // b
  gemm(B0, 0, wt_opwo, 0, F1, 0, (bf16*)0, 0, (const float*)0, 256, 4096, 32, 0, 1); // T[i][d*128+z]
  { size_t n = (size_t)256 * 4096; k_prep_T<<<EW(n)>>>(F1, B2, n); }                 // Tt[i][z][d]
  gemm(B1, 0, B2, 4096, F2, 32768, (bf16*)0, 0, opm_bo, 256, 128, 32, 0, 256);       // out[i][j][z]
  k_opm_final<<<EW(ZT)>>>(zW, F2, pm, ZT);

  // ======================= 4/5) Triangle multiply (out, in) =======================
  for (int t = 0; t < 2; ++t) {
    int tr = (t == 0) ? 0 : 1;   // outgoing: no transpose in pack; incoming: transpose
    ln(zW, tm[t].lng, tm[t].lnb, (float*)0, xbf, 65536, 128);
    gemm(xbf, 0, wt_tm[t][0], 0, F0, 0, (bf16*)0, 0, tm[t].bap, 65536, 128, 128, 0, 1);
    gemm(xbf, 0, wt_tm[t][1], 0, F1, 0, (bf16*)0, 0, tm[t].bag, 65536, 128, 128, 0, 1);
    k_gatemask<<<EW(ZT)>>>(F2, F0, F1, pm, 128, ZT);
    k_pack<<<EW(ZT)>>>(F2, B0, tr, ZT);                                 // A [c][i][k]
    gemm(xbf, 0, wt_tm[t][2], 0, F0, 0, (bf16*)0, 0, tm[t].bbp, 65536, 128, 128, 0, 1);
    gemm(xbf, 0, wt_tm[t][3], 0, F1, 0, (bf16*)0, 0, tm[t].bbg, 65536, 128, 128, 0, 1);
    k_gatemask<<<EW(ZT)>>>(F2, F0, F1, pm, 128, ZT);
    k_pack<<<EW(ZT)>>>(F2, B1, tr, ZT);                                 // Bt [c][j][k]
    gemm(B0, 65536, B1, 65536, E0, 65536, (bf16*)0, 0, (const float*)0, 256, 256, 256, 0, 128);
    k_unpack<<<EW(ZT)>>>(E0, F0, ZT);                                   // [i][j][c]
    ln(F0, tm[t].log, tm[t].lob, (float*)0, B2, 65536, 128);
    gemm(B2, 0, wt_tm[t][4], 0, F1, 0, (bf16*)0, 0, tm[t].bz, 65536, 128, 128, 0, 1);
    gemm(xbf, 0, wt_tm[t][5], 0, F2, 0, (bf16*)0, 0, tm[t].bg2, 65536, 128, 128, 0, 1);
    k_combine<<<EW(ZT)>>>(zW, F1, F2, ZT);
  }

  // ======================= 6/7) Triangle attention (start, end) =======================
  for (int t = 0; t < 2; ++t) {
    const float* src = zW;
    const float* pmu = pm;
    if (t == 1) { k_transpose_z<<<EW(ZT)>>>(zW, E0, ZT); src = E0; pmu = pmT; }
    ln(src, ta[t].lng, ta[t].lnb, (float*)0, xbf, 65536, 128);
    gemm(xbf, 0, wt_ta[t][0], 0, (float*)0, 0, B0, 0, (const float*)0, 65536, 128, 128, 0, 1); // q
    gemm(xbf, 0, wt_ta[t][1], 0, (float*)0, 0, B1, 0, (const float*)0, 65536, 128, 128, 0, 1); // k
    gemm(xbf, 0, wt_ta[t][2], 0, (float*)0, 0, B2, 0, (const float*)0, 65536, 128, 128, 0, 1); // v
    gemm(xbf, 0, wt_ta[t][3], 0, F1, 0, (bf16*)0, 0, ta[t].bg, 65536, 128, 128, 0, 1);         // gate
    gemm(xbf, 0, wt_tab[t], 0, F2, 0, (bf16*)0, 0, (const float*)0, 65536, 4, 128, 0, 1);      // tb raw
    k_bias_build<<<EW((size_t)4 * 65536)>>>(biasb, F2, 4, (size_t)4 * 65536);
    k_kbias<<<EW(65536)>>>(pmu, kb, 65536);
    attn(B0, B1, B2, biasb, kb, F0, 4, 256);
    k_gate_bf<<<EW(ZT)>>>(B3, F0, F1, ZT);
    gemm(B3, 0, wt_ta[t][4], 0, F2, 0, (bf16*)0, 0, ta[t].bo, 65536, 128, 128, 0, 1);
    if (t == 0) k_add<<<EW(ZT)>>>(zW, F2, ZT);
    else        k_add_T<<<EW(ZT)>>>(zW, F2, ZT);
  }

  // ======================= 8) Pair transition =======================
  ln(zW, pt_ln_g, pt_ln_b, (float*)0, xbf, 65536, 128);
  gemm(xbf, 0, wt_pt1, 0, (float*)0, 0, hbf, 0, pt_b1, 65536, 512, 128, 1, 1);
  gemm(hbf, 0, wt_pt2, 0, F0, 0, (bf16*)0, 0, pt_b2, 65536, 128, 512, 0, 1);
  k_add_rowmask<<<EW(ZT)>>>(zW, F0, pm, 128, ZT);

  // --- outputs: (m, z) concatenated ---
  (void)hipMemcpyAsync((float*)d_out, mW, MT * 4, hipMemcpyDeviceToDevice, stream);
  (void)hipMemcpyAsync((float*)d_out + MT, zW, ZT * 4, hipMemcpyDeviceToDevice, stream);
}
